// MixtureOfLinear_81252191306059
// MI455X (gfx1250) — compile-verified, matrix operand order checked by reading, not run on hardware
//
#include <hip/hip_runtime.h>
#include <hip/hip_bf16.h>
#include <math.h>
#include <cstdint>

// MoE-LoRA linear for MI455X (gfx1250, wave32).
// Roofline: 275 GFLOP base GEMM vs ~320MB HBM traffic (14us @23.3TB/s) ->
// heavily matrix-throughput bound in fp32. Convert once to bf16 and use
// V_WMMA_F32_16X16X32_BF16 (f32 accum); W_base bf16 (32MB) is L2-resident.
// LoRA up-projection (K=128) and bias are fused into the GEMM epilogue.
// LDS staging uses GLOBAL_LOAD_ASYNC_TO_LDS_B128 (ASYNCcnt) when available.
//
// Workspace requirement: ~105 MB (xb 64MB + wb 32MB + af/bf/gb/wr ~4.25MB).

typedef __bf16 bf16_t;
typedef __attribute__((ext_vector_type(16))) __bf16 v16bf;
typedef __attribute__((ext_vector_type(8)))  __bf16 v8bf;
typedef __attribute__((ext_vector_type(8)))  float  v8f;
typedef __attribute__((ext_vector_type(4)))  float  v4f;
typedef int v4i_vs __attribute__((vector_size(16)));   // matches builtin's pointee type

#define D_IN   4096
#define D_OUT  4096
#define NEXP   8
#define RANK   16
#define ER     128          // NEXP*RANK, flattened LoRA inner dim
#define SCALING 2.0f        // LORA_ALPHA / RANK

#define AS1 __attribute__((address_space(1)))
#define AS3 __attribute__((address_space(3)))

#if defined(__gfx1250__) && __has_builtin(__builtin_amdgcn_global_load_async_to_lds_b128)
#define USE_ASYNC_LDS 1
#else
#define USE_ASYNC_LDS 0
#endif

static __device__ __forceinline__ void async_wait_all() {
#if USE_ASYNC_LDS
#if __has_builtin(__builtin_amdgcn_s_wait_asynccnt)
  __builtin_amdgcn_s_wait_asynccnt(0);
#else
  asm volatile("s_wait_asynccnt 0x0" ::: "memory");
#endif
#endif
}

static __device__ __forceinline__ v8f vzero8() {
  v8f z = {0.f, 0.f, 0.f, 0.f, 0.f, 0.f, 0.f, 0.f};
  return z;
}

static __device__ __forceinline__ v16bf make_frag(v8bf lo, v8bf hi) {
  return __builtin_shufflevector(lo, hi, 0, 1, 2, 3, 4, 5, 6, 7,
                                         8, 9, 10, 11, 12, 13, 14, 15);
}

// Fragment for 16x16x32 bf16 WMMA from a row-major [rows x ld] bf16 matrix.
// CDNA5 dense 16-bit layout: lane holds row (lane&15); K chunks
// {k0..k0+7} in v0-3 and {k0+16..k0+23} in v4-7, k0 = (lane>=16)?8:0.
static __device__ __forceinline__ v16bf load_frag_g(const bf16_t* __restrict__ base,
                                                    int row, int ld, int kk, int lane) {
  int k0 = (lane >= 16) ? 8 : 0;
  const bf16_t* p = base + (long)row * ld + kk + k0;
  v8bf lo = *reinterpret_cast<const v8bf*>(p);
  v8bf hi = *reinterpret_cast<const v8bf*>(p + 16);
  return make_frag(lo, hi);
}

// ---------------------------------------------------------------------------
// K1: per-token (one wave32 per token): x -> bf16, router logits -> top-2
// softmax -> dense per-expert weight table wr[T,8] (zeros for unselected).
// ---------------------------------------------------------------------------
__global__ __launch_bounds__(256)
void moe_router_convert(const float* __restrict__ x,
                        const float* __restrict__ Wr,
                        bf16_t* __restrict__ xb,
                        float* __restrict__ wtab) {
  int lane = threadIdx.x & 31;
  int wave = threadIdx.x >> 5;
  int t = blockIdx.x * 8 + wave;
  const float* xrow = x + (long)t * D_IN;
  bf16_t* xbrow = xb + (long)t * D_IN;

  // pass 1: vectorized fp32 -> bf16 conversion (32B load / 16B store per lane)
  #pragma unroll
  for (int i = 0; i < D_IN / 256; ++i) {
    int d = i * 256 + lane * 8;
    v4f a0 = *reinterpret_cast<const v4f*>(xrow + d);
    v4f a1 = *reinterpret_cast<const v4f*>(xrow + d + 4);
    v8bf b;
    #pragma unroll
    for (int j = 0; j < 4; ++j) {
      b[j]     = (bf16_t)a0[j];
      b[j + 4] = (bf16_t)a1[j];
    }
    *reinterpret_cast<v8bf*>(xbrow + d) = b;
  }

  // pass 2: router logits (W_router is 128KB -> cache resident)
  float acc[NEXP];
  #pragma unroll
  for (int e = 0; e < NEXP; ++e) acc[e] = 0.f;
  for (int i = 0; i < D_IN / 32; ++i) {
    int d = i * 32 + lane;
    float xv = xrow[d];
    #pragma unroll
    for (int e = 0; e < NEXP; ++e) acc[e] += xv * Wr[e * D_IN + d];
  }
  #pragma unroll
  for (int e = 0; e < NEXP; ++e) {
    #pragma unroll
    for (int off = 16; off > 0; off >>= 1)
      acc[e] += __shfl_xor(acc[e], off, 32);
  }

  // top-2 + softmax over the two selected logits (all lanes redundantly)
  int e1 = 0;
  float l1 = acc[0];
  #pragma unroll
  for (int e = 1; e < NEXP; ++e)
    if (acc[e] > l1) { l1 = acc[e]; e1 = e; }
  int e2 = (e1 == 0) ? 1 : 0;
  float l2 = acc[e2];
  #pragma unroll
  for (int e = 0; e < NEXP; ++e)
    if (e != e1 && acc[e] > l2) { l2 = acc[e]; e2 = e; }
  float p1 = 1.f / (1.f + __expf(l2 - l1));   // weight of larger logit
  float p2 = 1.f - p1;
  if (lane < NEXP)
    wtab[(long)t * NEXP + lane] = (lane == e1) ? p1 : ((lane == e2) ? p2 : 0.f);
}

// ---------------------------------------------------------------------------
// K2a: bulk fp32 -> bf16 conversion (grid-stride, 8 elems / thread / iter)
// ---------------------------------------------------------------------------
__global__ __launch_bounds__(256)
void conv_f32_bf16(const float* __restrict__ src, bf16_t* __restrict__ dst, long n8) {
  long i = (long)blockIdx.x * blockDim.x + threadIdx.x;
  long stride = (long)gridDim.x * blockDim.x;
  for (; i < n8; i += stride) {
    long d = i * 8;
    v4f a0 = *reinterpret_cast<const v4f*>(src + d);
    v4f a1 = *reinterpret_cast<const v4f*>(src + d + 4);
    v8bf b;
    #pragma unroll
    for (int j = 0; j < 4; ++j) {
      b[j]     = (bf16_t)a0[j];
      b[j + 4] = (bf16_t)a1[j];
    }
    *reinterpret_cast<v8bf*>(dst + d) = b;
  }
}

// K2b: pack B [E, D_OUT, R] -> B_flat [D_OUT, E*R] bf16 (k index = e*16+r)
__global__ __launch_bounds__(256)
void pack_bflat(const float* __restrict__ Bsrc, bf16_t* __restrict__ bfm) {
  int idx = blockIdx.x * blockDim.x + threadIdx.x;
  if (idx >= D_OUT * ER) return;
  int o = idx >> 7;
  int j = idx & (ER - 1);
  int e = j >> 4;
  int r = j & (RANK - 1);
  bfm[idx] = (bf16_t)Bsrc[((long)e * D_OUT + o) * RANK + r];
}

// ---------------------------------------------------------------------------
// K3: g[t, e*16+r] = 2 * w[t,e] * sum_d x[t,d]*A[e,r,d]   (bf16 out)
// One wave per 16-token strip, N = 128 (A_flat is 1MB -> L2 resident).
// ---------------------------------------------------------------------------
__global__ __launch_bounds__(256)
void moe_lora_down(const bf16_t* __restrict__ xb,
                   const bf16_t* __restrict__ af,
                   const float* __restrict__ wtab,
                   bf16_t* __restrict__ gb) {
  int lane = threadIdx.x & 31;
  int wave = threadIdx.x >> 5;
  int tBase = blockIdx.x * 128 + wave * 16;
  int col = lane & 15;
  int hi8 = (lane >= 16) ? 8 : 0;

  v8f acc[8];
  #pragma unroll
  for (int n = 0; n < 8; ++n) acc[n] = vzero8();

  for (int kk = 0; kk < D_IN; kk += 32) {
    v16bf a = load_frag_g(xb, tBase + col, D_IN, kk, lane);
    #pragma unroll
    for (int n = 0; n < 8; ++n) {
      v16bf b = load_frag_g(af, n * 16 + col, D_IN, kk, lane);
      acc[n] = __builtin_amdgcn_wmma_f32_16x16x32_bf16(
          false, a, false, b, (short)0, acc[n], false, false);
    }
  }

  // lane holds D elems (m = r + hi8, n-col = col); j>>4 == n -> expert = n
  #pragma unroll
  for (int n = 0; n < 8; ++n) {
    #pragma unroll
    for (int r = 0; r < 8; ++r) {
      int t = tBase + r + hi8;
      float wgt = SCALING * wtab[(long)t * NEXP + n];
      gb[(long)t * ER + n * 16 + col] = (bf16_t)(acc[n][r] * wgt);
    }
  }
}

// ---------------------------------------------------------------------------
// K4: out = xb @ wbᵀ + bias + gb @ B_flatᵀ
// 128x128 block tile, 8 waves (2x4), 32-wide K steps, double-buffered LDS
// filled by GLOBAL_LOAD_ASYNC_TO_LDS_B128 (no dest VGPRs, ASYNCcnt-tracked).
// LDS row pitch = 40 halves (80B): the 16 fragment rows land on disjoint
// bank quads -> conflict-free ds_load_b128.
// ---------------------------------------------------------------------------
#define BM 128
#define BN 128
#define BK 32
#define LDP 40   // LDS row pitch in bf16 elements

__global__ __launch_bounds__(256)
void moe_base_gemm(const bf16_t* __restrict__ xb,
                   const bf16_t* __restrict__ wb,
                   const bf16_t* __restrict__ gb,
                   const bf16_t* __restrict__ bfm,
                   const float* __restrict__ bias,
                   float* __restrict__ out) {
  __shared__ __attribute__((aligned(16))) bf16_t lA[2][BM * LDP];
  __shared__ __attribute__((aligned(16))) bf16_t lB[2][BN * LDP];

  int tid  = threadIdx.x;
  int lane = tid & 31;
  int wave = tid >> 5;
  int wm = wave >> 2;              // 0..1 -> 64-row strip
  int wn = wave & 3;               // 0..3 -> 32-col strip
  int tBase = blockIdx.y * BM;
  int nBase = blockIdx.x * BN;
  int col = lane & 15;
  int k0  = (lane >= 16) ? 8 : 0;
  int hi8 = k0;

  auto stage = [&](int buf, int kk) {
    #pragma unroll
    for (int it = 0; it < 2; ++it) {
      int idx = tid + it * 256;        // 0..511 -> 128 rows x 4 chunks
      int r = idx >> 2;
      int c = (idx & 3) * 8;
      const bf16_t* pa = xb + (long)(tBase + r) * D_IN + kk + c;
      const bf16_t* pb = wb + (long)(nBase + r) * D_IN + kk + c;
      bf16_t* la = &lA[buf][r * LDP + c];
      bf16_t* lb = &lB[buf][r * LDP + c];
      if (kk + 2 * BK < D_IN) {        // deep prefetch -> global_prefetch_b8
        __builtin_prefetch(pa + 2 * BK, 0, 1);
        __builtin_prefetch(pb + 2 * BK, 0, 1);
      }
#if USE_ASYNC_LDS
      // 16B/lane straight to LDS, no destination VGPRs, tracked by ASYNCcnt.
      __builtin_amdgcn_global_load_async_to_lds_b128(
          (AS1 v4i_vs*)reinterpret_cast<uintptr_t>(pa),
          (AS3 v4i_vs*)reinterpret_cast<uintptr_t>(la), 0, 0);
      __builtin_amdgcn_global_load_async_to_lds_b128(
          (AS1 v4i_vs*)reinterpret_cast<uintptr_t>(pb),
          (AS3 v4i_vs*)reinterpret_cast<uintptr_t>(lb), 0, 0);
#else
      *reinterpret_cast<v8bf*>(la) = *reinterpret_cast<const v8bf*>(pa);
      *reinterpret_cast<v8bf*>(lb) = *reinterpret_cast<const v8bf*>(pb);
#endif
    }
  };

  v8f acc[4][2];
  #pragma unroll
  for (int mi = 0; mi < 4; ++mi)
    #pragma unroll
    for (int ni = 0; ni < 2; ++ni) acc[mi][ni] = vzero8();

  stage(0, 0);
  async_wait_all();
  __syncthreads();

  for (int kk = 0; kk < D_IN; kk += BK) {
    int buf = (kk >> 5) & 1;
    if (kk + BK < D_IN) stage(buf ^ 1, kk + BK);   // overlaps with WMMAs below

    v16bf afr[4], bfr[2];
    #pragma unroll
    for (int mi = 0; mi < 4; ++mi) {
      const bf16_t* p = &lA[buf][(wm * 64 + mi * 16 + col) * LDP + k0];
      afr[mi] = make_frag(*reinterpret_cast<const v8bf*>(p),
                          *reinterpret_cast<const v8bf*>(p + 16));
    }
    #pragma unroll
    for (int ni = 0; ni < 2; ++ni) {
      const bf16_t* p = &lB[buf][(wn * 32 + ni * 16 + col) * LDP + k0];
      bfr[ni] = make_frag(*reinterpret_cast<const v8bf*>(p),
                          *reinterpret_cast<const v8bf*>(p + 16));
    }
    #pragma unroll
    for (int mi = 0; mi < 4; ++mi)
      #pragma unroll
      for (int ni = 0; ni < 2; ++ni)
        acc[mi][ni] = __builtin_amdgcn_wmma_f32_16x16x32_bf16(
            false, afr[mi], false, bfr[ni], (short)0, acc[mi][ni], false, false);

    async_wait_all();
    __syncthreads();
  }

  // Fused LoRA up-projection: 4 more K-steps over [T,128] x [D_OUT,128]
  #pragma unroll
  for (int ks = 0; ks < ER; ks += 32) {
    v16bf afr[4], bfr[2];
    #pragma unroll
    for (int mi = 0; mi < 4; ++mi)
      afr[mi] = load_frag_g(gb, tBase + wm * 64 + mi * 16 + col, ER, ks, lane);
    #pragma unroll
    for (int ni = 0; ni < 2; ++ni)
      bfr[ni] = load_frag_g(bfm, nBase + wn * 32 + ni * 16 + col, ER, ks, lane);
    #pragma unroll
    for (int mi = 0; mi < 4; ++mi)
      #pragma unroll
      for (int ni = 0; ni < 2; ++ni)
        acc[mi][ni] = __builtin_amdgcn_wmma_f32_16x16x32_bf16(
            false, afr[mi], false, bfr[ni], (short)0, acc[mi][ni], false, false);
  }

  // bias + store (C/D layout: lane -> column col, VGPR r -> row r + hi8)
  #pragma unroll
  for (int ni = 0; ni < 2; ++ni) {
    int o = nBase + wn * 32 + ni * 16 + col;
    float bv = bias[o];
    #pragma unroll
    for (int mi = 0; mi < 4; ++mi) {
      int trow = tBase + wm * 64 + mi * 16 + hi8;
      #pragma unroll
      for (int r = 0; r < 8; ++r)
        out[(long)(trow + r) * D_OUT + o] = acc[mi][ni][r] + bv;
    }
  }
}

// ---------------------------------------------------------------------------
extern "C" void kernel_launch(void* const* d_in, const int* in_sizes, int n_in,
                              void* d_out, int out_size, void* d_ws, size_t ws_size,
                              hipStream_t stream) {
  const float* x  = (const float*)d_in[0];
  const float* Wb = (const float*)d_in[1];
  const float* bb = (const float*)d_in[2];
  const float* Wr = (const float*)d_in[3];
  const float* A  = (const float*)d_in[4];
  const float* Bm = (const float*)d_in[5];
  float* out = (float*)d_out;
  const int T = in_sizes[0] / D_IN;   // 8192

  char* ws = (char*)d_ws;
  bf16_t* xb  = (bf16_t*)ws;  ws += (size_t)T * D_IN * 2;       // 64 MB
  bf16_t* wbb = (bf16_t*)ws;  ws += (size_t)D_OUT * D_IN * 2;   // 32 MB
  bf16_t* af  = (bf16_t*)ws;  ws += (size_t)ER * D_IN * 2;      // 1 MB
  bf16_t* bfm = (bf16_t*)ws;  ws += (size_t)D_OUT * ER * 2;     // 1 MB
  bf16_t* gb  = (bf16_t*)ws;  ws += (size_t)T * ER * 2;         // 2 MB
  float*  wt  = (float*)ws;   ws += (size_t)T * NEXP * 4;       // 256 KB

  moe_router_convert<<<T / 8, 256, 0, stream>>>(x, Wr, xb, wt);
  conv_f32_bf16<<<2048, 256, 0, stream>>>(Wb, wbb, (long)D_OUT * D_IN / 8);
  conv_f32_bf16<<<256, 256, 0, stream>>>(A, af, (long)ER * D_IN / 8);
  pack_bflat<<<(D_OUT * ER + 255) / 256, 256, 0, stream>>>(Bm, bfm);
  moe_lora_down<<<T / 128, 256, 0, stream>>>(xb, af, wt, gb);
  dim3 g(D_OUT / BN, T / BM);
  moe_base_gemm<<<g, 256, 0, stream>>>(xb, wbb, gb, bfm, bb, out);
}
